// GraphNeuralPDESolver_31928786878525
// MI455X (gfx1250) — matrix operand order repaired and checked by reading, not compiled, not run on hardware
//
#include <hip/hip_runtime.h>
#include <hip/hip_bf16.h>

typedef _Float16 v16h __attribute__((ext_vector_type(16)));
typedef float    v8f  __attribute__((ext_vector_type(8)));
typedef _Float16 h2v  __attribute__((ext_vector_type(2)));

#define B_   16
#define NG_  4096
#define NM_  2048
#define NEA_ 8192   // grid2mesh / mesh2grid edges
#define NEM_ 7936   // multimesh edges: 2*(2048+1024+512+256+128)

// ---------------------------------------------------------------------------
// Argument block for the fused 3-layer MLP kernel
// ---------------------------------------------------------------------------
struct MlpArgs {
  const _Float16* Wf1; const float* b1;
  const _Float16* Wf2; const float* b2;
  const _Float16* Wf3; const float* b3;
  const float* ln_s; const float* ln_b;
  const _Float16* src_main;              // latent being updated (residual src)
  const _Float16* srcA; const int* idxA; // gather source A
  const _Float16* srcB; const int* idxB; // gather source B
  const float* raw0; const float* raw1; const float* raw2; // raw f32 features
  _Float16* out;
  float* out_f32;
};

// swish via fast hardware reciprocal: x * rcp(1+exp(-x))
// (avoids the IEEE div_scale/rcp/fma Newton sequence in the GEMM epilogue)
__device__ __forceinline__ float swishf(float x) {
  return x * __builtin_amdgcn_rcpf(1.0f + __expf(-x));
}

// One wave computes a 16x16 D tile pair (rt, ct0) and (rt, ct0+1).
// S: LDS activations, row-major [32][KT*32] f16.
// Wf: weight fragments, layout ((kt*8+ct)*32 + lane)*16 + half.
template<int KT>
__device__ __forceinline__ void gemm2(const _Float16* S, const _Float16* Wf,
                                      v8f& acc0, v8f& acc1,
                                      int rt, int ct0, int lane) {
  const int m  = lane & 15;
  const int hi = lane >> 4;
  const int W  = KT * 32;
  #pragma unroll
  for (int kt = 0; kt < KT; ++kt) {
    v16h a;
    #pragma unroll
    for (int e = 0; e < 8; ++e) {
      // 16-bit A 16x32 layout (matches B frags): VGPR e holds K = hi*16 + e*2, +1
      int K = kt * 32 + hi * 16 + (e << 1);
      h2v p = *(const h2v*)(S + (rt * 16 + m) * W + K);
      a[2 * e]     = p.x;
      a[2 * e + 1] = p.y;
    }
    v16h b0 = *(const v16h*)(Wf + (((size_t)(kt * 8 + ct0    ) * 32 + lane) * 16));
    v16h b1 = *(const v16h*)(Wf + (((size_t)(kt * 8 + ct0 + 1) * 32 + lane) * 16));
    acc0 = __builtin_amdgcn_wmma_f32_16x16x32_f16(false, a, false, b0, (short)0, acc0, false, false);
    acc1 = __builtin_amdgcn_wmma_f32_16x16x32_f16(false, a, false, b1, (short)0, acc1, false, false);
  }
}

// D tile store: lane holds N=lane%16, rows rt*16 + (lane>=16?8:0) + r
__device__ __forceinline__ void store_h(v8f acc, const float* bias, _Float16* H,
                                        int rt, int ct, int lane, bool act) {
  int hi = lane >> 4, col = ct * 16 + (lane & 15);
  float bv = bias[col];
  #pragma unroll
  for (int r = 0; r < 8; ++r) {
    int row = rt * 16 + hi * 8 + r;
    float x = acc[r] + bv;
    if (act) x = swishf(x);
    H[row * 128 + col] = (_Float16)x;
  }
}

__device__ __forceinline__ void store_y(v8f acc, const float* bias, float* Y,
                                        int rt, int ct, int lane) {
  int hi = lane >> 4, col = ct * 16 + (lane & 15);
  float bv = bias[col];
  #pragma unroll
  for (int r = 0; r < 8; ++r) {
    Y[(rt * 16 + hi * 8 + r) * 128 + col] = acc[r] + bv;
  }
}

// ---------------------------------------------------------------------------
// MODE: 0 embed_grid  1 embed_mesh  2 embed_edge (K padded to 32)
//       3 edge block (K=384, gather el + nodeA[idxA] + nodeB[idxB])
//       4 g2m mesh_node (K=256, agg = 4 contiguous edges)
//       5 mesh node    (K=256, analytic multimesh agg)
//       6 m2g grid_node(K=256, analytic 2-edge agg)
//       7 grid self    (K=128)      8 decoder (K=128, dout=1, +u residual)
// ---------------------------------------------------------------------------
template<int MODE>
__global__ void __launch_bounds__(256)
mlp_kernel(MlpArgs A) {
  constexpr int KINPAD = (MODE < 3) ? 32 : (MODE == 3) ? 384 : (MODE <= 6) ? 256 : 128;
  constexpr int KT1 = KINPAD / 32;
  constexpr bool RES = (MODE >= 3 && MODE <= 7);

  __shared__ __align__(16) _Float16 sm0[12288];     // 24KB: X, later h2(8KB)+Y(16KB)
  __shared__ __align__(16) _Float16 smH1[32 * 128]; // 8KB
  _Float16* X  = sm0;
  _Float16* H2 = sm0;
  float*    Y  = (float*)(sm0 + 4096);

  const int tid = threadIdx.x;
  const int rowBase = blockIdx.x * 32;

  // ---- stage X (f32 -> f16) with fused gather / aggregation ----
  for (int idx = tid; idx < 32 * KINPAD; idx += 256) {
    int row = idx / KINPAD;
    int k   = idx - row * KINPAD;
    int grow = rowBase + row;
    int ent = grow >> 4, b = grow & 15;
    float v = 0.f;
    if constexpr (MODE == 0) {
      if (k < 4)       v = A.raw0[b * 4 + k];                 // specs
      else if (k == 4) v = A.raw1[b * NG_ + ent];             // u_inp
      else if (k < 7)  v = A.raw2[ent * 2 + (k - 5)];         // grid_static
    } else if constexpr (MODE == 1) {
      if (k >= 5 && k < 7) v = A.raw2[ent * 2 + (k - 5)];     // mesh_static
    } else if constexpr (MODE == 2) {
      if (k == 0) v = A.raw0[ent];                            // edge feature
    } else if constexpr (MODE == 3) {
      if (k < 128)      v = (float)A.src_main[(size_t)grow * 128 + k];
      else if (k < 256) v = (float)A.srcA[((size_t)A.idxA[ent] * B_ + b) * 128 + (k - 128)];
      else              v = (float)A.srcB[((size_t)A.idxB[ent] * B_ + b) * 128 + (k - 256)];
    } else if constexpr (MODE == 4) {
      if (k < 128) v = (float)A.src_main[(size_t)grow * 128 + k];
      else {
        int kk = k - 128; float s = 0.f;
        #pragma unroll
        for (int c = 0; c < 4; ++c)
          s += (float)A.srcA[(((size_t)(4 * ent + c)) * B_ + b) * 128 + kk];
        v = s;
      }
    } else if constexpr (MODE == 5) {
      if (k < 128) v = (float)A.src_main[(size_t)grow * 128 + k];
      else {
        int kk = k - 128; float s = 0.f;
        constexpr int OFFP[5] = {0, 4096, 6144, 7168, 7680};
        #pragma unroll
        for (int p = 0; p < 5; ++p) {
          int d = 1 << p;
          if ((ent & (d - 1)) == 0) {
            int e1 = OFFP[p] + (((ent + NM_ - d) & (NM_ - 1)) >> p);
            int e2 = OFFP[p] + (NM_ >> p) + (ent >> p);
            s += (float)A.srcA[((size_t)e1 * B_ + b) * 128 + kk]
               + (float)A.srcA[((size_t)e2 * B_ + b) * 128 + kk];
          }
        }
        v = s;
      }
    } else if constexpr (MODE == 6) {
      if (k < 128) v = (float)A.src_main[(size_t)grow * 128 + k];
      else {
        int kk = k - 128;
        int c  = ent & 1;
        int e1 = 4 * ((ent - c) >> 1) + c;
        int c2 = c + 2;
        int m2 = ((ent - c2 + NG_) >> 1) & (NM_ - 1);
        int e2 = 4 * m2 + c2;
        v = (float)A.srcA[((size_t)e1 * B_ + b) * 128 + kk]
          + (float)A.srcA[((size_t)e2 * B_ + b) * 128 + kk];
      }
    } else { // 7, 8
      v = (float)A.src_main[(size_t)grow * 128 + k];
    }
    X[row * KINPAD + k] = (_Float16)v;
  }
  __syncthreads();

  const int lane = tid & 31;
  const int wave = tid >> 5;
  const int rt   = wave >> 2;        // 0..1 (row tile)
  const int ct0  = (wave & 3) * 2;   // 0,2,4,6 (col tile base)

  { // layer 1: X @ W1 -> swish -> H1
    v8f a0 = {}; v8f a1 = {};
    gemm2<KT1>(X, A.Wf1, a0, a1, rt, ct0, lane);
    store_h(a0, A.b1, smH1, rt, ct0,     lane, true);
    store_h(a1, A.b1, smH1, rt, ct0 + 1, lane, true);
  }
  __syncthreads();
  { // layer 2: H1 @ W2 -> swish -> H2 (reuses X region; X fully consumed)
    v8f a0 = {}; v8f a1 = {};
    gemm2<4>(smH1, A.Wf2, a0, a1, rt, ct0, lane);
    store_h(a0, A.b2, H2, rt, ct0,     lane, true);
    store_h(a1, A.b2, H2, rt, ct0 + 1, lane, true);
  }
  __syncthreads();

  if constexpr (MODE == 8) {
    // decoder layer 3: 128 -> 1 dot with raw f32 W3, + b3, + u_inp residual
    int row = tid >> 3, seg = tid & 7;
    float s = 0.f;
    #pragma unroll
    for (int c = 0; c < 16; ++c) {
      int kk = seg * 16 + c;
      s += (float)H2[row * 128 + kk] * A.raw0[kk];
    }
    s += __shfl_xor(s, 1, 32);
    s += __shfl_xor(s, 2, 32);
    s += __shfl_xor(s, 4, 32);
    if (seg == 0) {
      int grow = rowBase + row;
      int g = grow >> 4, b = grow & 15;
      A.out_f32[b * NG_ + g] = A.raw1[b * NG_ + g] + s + A.b3[0];
    }
  } else {
    { // layer 3: H2 @ W3 + b3 -> Y (f32, LDS)
      v8f a0 = {}; v8f a1 = {};
      gemm2<4>(H2, A.Wf3, a0, a1, rt, ct0, lane);
      store_y(a0, A.b3, Y, rt, ct0,     lane);
      store_y(a1, A.b3, Y, rt, ct0 + 1, lane);
    }
    __syncthreads();
    // LayerNorm (+ optional residual), 8 lanes per row
    int row = tid >> 3, seg = tid & 7;
    float s = 0.f, s2 = 0.f;
    float vals[16];
    #pragma unroll
    for (int c = 0; c < 16; ++c) {
      float y = Y[row * 128 + seg * 16 + c];
      vals[c] = y; s += y; s2 += y * y;
    }
    #pragma unroll
    for (int msk = 1; msk < 8; msk <<= 1) {
      s  += __shfl_xor(s,  msk, 32);
      s2 += __shfl_xor(s2, msk, 32);
    }
    float mu  = s  * (1.f / 128.f);
    float var = s2 * (1.f / 128.f) - mu * mu;
    float inv = rsqrtf(var + 1e-5f);
    int grow = rowBase + row;
    #pragma unroll
    for (int c = 0; c < 16; ++c) {
      int col = seg * 16 + c;
      float o = (vals[c] - mu) * inv * A.ln_s[col] + A.ln_b[col];
      if constexpr (RES) o += (float)A.src_main[(size_t)grow * 128 + col];
      A.out[(size_t)grow * 128 + col] = (_Float16)o;
    }
  }
}

// ---------------------------------------------------------------------------
// Weight prep: f32 [K,128] -> f16 WMMA B-fragments (K padded to mult. of 32)
// frag[idx], idx = ((kt*8+ct)*32 + lane)*16 + half ; k = kt*32 + 16*(lane>=16)+half
// ---------------------------------------------------------------------------
struct PrepDesc { const float* src; _Float16* dst; int kin; int kt; };
struct PrepArgs { int count; PrepDesc d[72]; };

__global__ void __launch_bounds__(256)
prep_kernel(PrepArgs A) {
  int m = blockIdx.y;
  if (m >= A.count) return;
  PrepDesc D = A.d[m];
  int total = D.kt * 4096;
  for (int idx = blockIdx.x * 256 + threadIdx.x; idx < total; idx += gridDim.x * 256) {
    int half = idx & 15;
    int lane = (idx >> 4) & 31;
    int ct   = (idx >> 9) & 7;
    int kt   = idx >> 12;
    int k = kt * 32 + ((lane >> 4) << 4) + half;
    int n = ct * 16 + (lane & 15);
    float v = (k < D.kin) ? D.src[k * 128 + n] : 0.f;
    D.dst[idx] = (_Float16)v;
  }
}

// ---------------------------------------------------------------------------
// Host orchestration
// ---------------------------------------------------------------------------
extern "C" void kernel_launch(void* const* d_in, const int* in_sizes, int n_in,
                              void* d_out, int out_size, void* d_ws, size_t ws_size,
                              hipStream_t stream) {
  (void)in_sizes; (void)n_in; (void)out_size; (void)ws_size;
  auto F = [&](int i) { return (const float*)d_in[i]; };
  auto I = [&](int i) { return (const int*)d_in[i]; };

  char* ws = (char*)d_ws;
  size_t off = 0;
  auto alloc = [&](size_t bytes) -> void* {
    off = (off + 63) & ~(size_t)63;
    void* p = ws + off;
    off += bytes;
    return p;
  };

  const size_t RG = (size_t)NG_ * B_, RM = (size_t)NM_ * B_;
  const size_t REA = (size_t)NEA_ * B_, REM = (size_t)NEM_ * B_;
  _Float16* gl  = (_Float16*)alloc(RG  * 128 * 2);  // grid latent
  _Float16* ml  = (_Float16*)alloc(RM  * 128 * 2);  // mesh latent
  _Float16* ela = (_Float16*)alloc(REA * 128 * 2);  // g2m/m2g edge latent
  _Float16* elm = (_Float16*)alloc(REM * 128 * 2);  // multimesh edge latent

  PrepArgs pa; pa.count = 0;
  auto frag = [&](int widx, int kin) -> _Float16* {
    int kt = (kin + 31) / 32;
    _Float16* dst = (_Float16*)alloc((size_t)kt * 4096 * 2);
    pa.d[pa.count++] = PrepDesc{F(widx), dst, kin, kt};
    return dst;
  };

  struct BW { const float *lns, *lnb, *b1, *b2, *b3; _Float16 *w1, *w2, *w3; };
  auto mkblk = [&](int base, int kin) -> BW {
    BW w;
    w.lns = F(base);     w.lnb = F(base + 1);
    w.w1 = frag(base + 2, kin); w.b1 = F(base + 3);
    w.w2 = frag(base + 4, 128); w.b2 = F(base + 5);
    w.w3 = frag(base + 6, 128); w.b3 = F(base + 7);
    return w;
  };

  // params pytree (sorted dict keys), leaves start at d_in[7]
  BW g2m_edge  = mkblk(7,  384);
  BW g2m_eedge = mkblk(15, 1);
  BW g2m_egrid = mkblk(23, 7);
  BW g2m_emesh = mkblk(31, 7);
  BW g2m_gnode = mkblk(39, 128);
  BW g2m_mnode = mkblk(47, 256);
  _Float16* dec_w1 = frag(55, 128);                // m2g.decoder (no LN)
  _Float16* dec_w2 = frag(57, 128);
  BW m2g_edge   = mkblk(61, 384);
  BW m2g_eedge  = mkblk(69, 1);
  BW m2g_gnode  = mkblk(77, 256);
  BW mesh_eedge = mkblk(85, 1);
  BW mesh_edge[6], mesh_node[6];
  for (int s = 0; s < 6; ++s) {
    mesh_edge[s] = mkblk(93 + 16 * s,     384);
    mesh_node[s] = mkblk(93 + 16 * s + 8, 256);
  }
  const int* g2m_s = I(189);
  const int* g2m_r = I(190);
  const int* ms    = I(191);
  const int* mr    = I(192);

  prep_kernel<<<dim3(8, (unsigned)pa.count), dim3(256), 0, stream>>>(pa);

  auto base_args = [&](const BW& w) {
    MlpArgs a{};
    a.Wf1 = w.w1; a.b1 = w.b1; a.Wf2 = w.w2; a.b2 = w.b2; a.Wf3 = w.w3; a.b3 = w.b3;
    a.ln_s = w.lns; a.ln_b = w.lnb;
    return a;
  };

  // ---- grid2mesh ----
  { MlpArgs a = base_args(g2m_egrid); a.raw0 = F(0); a.raw1 = F(1); a.raw2 = F(2); a.out = gl;
    mlp_kernel<0><<<(unsigned)(RG / 32), 256, 0, stream>>>(a); }
  { MlpArgs a = base_args(g2m_emesh); a.raw2 = F(3); a.out = ml;
    mlp_kernel<1><<<(unsigned)(RM / 32), 256, 0, stream>>>(a); }
  { MlpArgs a = base_args(g2m_eedge); a.raw0 = F(4); a.out = ela;
    mlp_kernel<2><<<(unsigned)(REA / 32), 256, 0, stream>>>(a); }
  { MlpArgs a = base_args(g2m_edge); a.src_main = ela;
    a.srcA = gl; a.idxA = g2m_s; a.srcB = ml; a.idxB = g2m_r; a.out = ela;
    mlp_kernel<3><<<(unsigned)(REA / 32), 256, 0, stream>>>(a); }
  { MlpArgs a = base_args(g2m_mnode); a.src_main = ml; a.srcA = ela; a.out = ml;
    mlp_kernel<4><<<(unsigned)(RM / 32), 256, 0, stream>>>(a); }
  { MlpArgs a = base_args(g2m_gnode); a.src_main = gl; a.out = gl;
    mlp_kernel<7><<<(unsigned)(RG / 32), 256, 0, stream>>>(a); }

  // ---- mesh processor (6 steps) ----
  { MlpArgs a = base_args(mesh_eedge); a.raw0 = F(5); a.out = elm;
    mlp_kernel<2><<<(unsigned)(REM / 32), 256, 0, stream>>>(a); }
  for (int s = 0; s < 6; ++s) {
    { MlpArgs a = base_args(mesh_edge[s]); a.src_main = elm;
      a.srcA = ml; a.idxA = ms; a.srcB = ml; a.idxB = mr; a.out = elm;
      mlp_kernel<3><<<(unsigned)(REM / 32), 256, 0, stream>>>(a); }
    { MlpArgs a = base_args(mesh_node[s]); a.src_main = ml; a.srcA = elm; a.out = ml;
      mlp_kernel<5><<<(unsigned)(RM / 32), 256, 0, stream>>>(a); }
  }

  // ---- mesh2grid + decoder ----
  { MlpArgs a = base_args(m2g_eedge); a.raw0 = F(6); a.out = ela;
    mlp_kernel<2><<<(unsigned)(REA / 32), 256, 0, stream>>>(a); }
  { MlpArgs a = base_args(m2g_edge); a.src_main = ela;
    a.srcA = ml; a.idxA = g2m_r; a.srcB = gl; a.idxB = g2m_s; a.out = ela;
    mlp_kernel<3><<<(unsigned)(REA / 32), 256, 0, stream>>>(a); }
  { MlpArgs a = base_args(m2g_gnode); a.src_main = gl; a.srcA = ela; a.out = gl;
    mlp_kernel<6><<<(unsigned)(RG / 32), 256, 0, stream>>>(a); }
  { MlpArgs a{};
    a.Wf1 = dec_w1; a.b1 = F(56); a.Wf2 = dec_w2; a.b2 = F(58);
    a.raw0 = F(59); a.b3 = F(60); a.raw1 = F(1);
    a.src_main = gl; a.out_f32 = (float*)d_out;
    mlp_kernel<8><<<(unsigned)(RG / 32), 256, 0, stream>>>(a); }
}
